// EncoderBlock_34213709480320
// MI455X (gfx1250) — matrix operand order repaired and checked by reading, not compiled, not run on hardware
//
#include <hip/hip_runtime.h>
#include <hip/hip_bf16.h>
#include <math.h>

// ---------------------------------------------------------------------------
// MI455X (gfx1250) encoder block.
//  - All large GEMMs use v_wmma_f32_16x16x32_bf16 (wave32, 16x16x32, f32 acc).
//  - Elementwise/bandwidth stages (PE, norm, depthwise conv, softmax) in f32.
//  - Scores are stored transposed so softmax over the query axis (m) is a
//    contiguous row reduction, and the V*A GEMM reads the attention matrix
//    as a direct [N][K] tile.
// ---------------------------------------------------------------------------

typedef __bf16 bf16;
typedef __bf16 v16bf __attribute__((ext_vector_type(16)));
typedef __bf16 v8bf  __attribute__((ext_vector_type(8)));
typedef float  v8f   __attribute__((ext_vector_type(8)));

#define KT  64   // K tile (2 wmma k-steps)
#define LDW 72   // padded LDS row stride (bf16 elems) -> 144B, 16B aligned

enum { EP_RELU_RES = 0, EP_RES = 1, EP_BF16 = 2, EP_SCALED_T = 3 };

struct GemmP {
  const bf16* A; const bf16* B; const float* Res;
  float* Of; bf16* Ob;
  long sA, sB, sO, sR;          // per-batch strides (elements)
  int  zA, zB, zO, zR;          // batch-index bases
  int  mA, mB, mO, mR;          // batch-index modulus (for broadcast/wrap)
  int  ldA, ldB, ldO;           // leading dims (elements)
  int  Kdim; float scale; int headOut;
};

union FragU { v16bf v; v8bf h[2]; };

// Tiled batched GEMM: Out = A(MxK) * B(KxN) with fused epilogues.
// TRANSA=1: A supplied as [K][M] row-major (transposed on LDS load).
// BNK=1:    B supplied as [N][K] row-major (direct copy), else [K][N].
template<int MT, int NT, int WM, int WN, int TRANSA, int BNK, int EP>
__global__ __launch_bounds__(256) void gemm_k(GemmP p) {
  __shared__ bf16 shA[MT * LDW];
  __shared__ bf16 shB[NT * LDW];

  const int tid  = threadIdx.x;
  const int lane = tid & 31;
  const int wave = tid >> 5;
  const int wm   = wave % WM;          // wave row (32 rows each)
  const int wn   = wave / WM;          // wave col (64 cols each)
  const int l15  = lane & 15;
  const int hi   = lane >> 4;          // half-wave select

  const int bz = blockIdx.z;
  const long offA = (long)((p.zA + bz) % p.mA) * p.sA;
  const long offB = (long)((p.zB + bz) % p.mB) * p.sB;
  long offO;
  if (p.headOut) {                      // scatter head z=(h*16+b) into [B,C,L]
    int zg = p.zO + bz;
    offO = ((long)((zg & 15) * 512 + (zg >> 4) * 64)) * 1024;
  } else {
    offO = (long)((p.zO + bz) % p.mO) * p.sO;
  }
  const long offR = (long)((p.zR + bz) % p.mR) * p.sR;

  const bf16* gA = p.A + offA;
  const bf16* gB = p.B + offB;

  const int m0 = blockIdx.y * MT;
  const int n0 = blockIdx.x * NT;

  v8f acc[2][4];
  const v8f vzero = {0.f, 0.f, 0.f, 0.f, 0.f, 0.f, 0.f, 0.f};
#pragma unroll
  for (int mf = 0; mf < 2; ++mf)
#pragma unroll
    for (int nf = 0; nf < 4; ++nf) acc[mf][nf] = vzero;

  for (int k0 = 0; k0 < p.Kdim; k0 += KT) {
    // hint-prefetch next K tile while we wait on the barrier
    if (k0 + KT < p.Kdim) {
      __builtin_prefetch(&gA[(long)(TRANSA ? (k0 + KT) : m0) * p.ldA], 0, 1);
      __builtin_prefetch(&gB[(long)(BNK ? n0 : (k0 + KT)) * p.ldB], 0, 1);
    }
    __syncthreads();
    // ---- stage A tile into LDS as [M][K] ----
    if (!TRANSA) {
#pragma unroll
      for (int pi = 0; pi < MT / 32; ++pi) {
        int q  = tid + pi * 256;
        int r  = q >> 3;
        int kk = (q & 7) * 8;
        *(v8bf*)(&shA[r * LDW + kk]) =
            *(const v8bf*)(&gA[(long)(m0 + r) * p.ldA + k0 + kk]);
      }
    } else {  // MT == 128 only
#pragma unroll
      for (int j = 0; j < 4; ++j) {
        int ml = (tid & 15) * 8;
        int kk = (tid >> 4) + j * 16;
        v8bf t = *(const v8bf*)(&gA[(long)(k0 + kk) * p.ldA + m0 + ml]);
#pragma unroll
        for (int e = 0; e < 8; ++e) shA[(ml + e) * LDW + kk] = t[e];
      }
    }
    // ---- stage B tile into LDS as [N][K] ----
    if (!BNK) {
      constexpr int R = NT / 8;
#pragma unroll
      for (int j = 0; j < (KT * R) / 256; ++j) {
        int nl = (tid % R) * 8;
        int kk = (tid / R) + j * (256 / R);
        v8bf t = *(const v8bf*)(&gB[(long)(k0 + kk) * p.ldB + n0 + nl]);
#pragma unroll
        for (int e = 0; e < 8; ++e) shB[(nl + e) * LDW + kk] = t[e];
      }
    } else {
#pragma unroll
      for (int pi = 0; pi < NT / 32; ++pi) {
        int q  = tid + pi * 256;
        int r  = q >> 3;
        int kk = (q & 7) * 8;
        *(v8bf*)(&shB[r * LDW + kk]) =
            *(const v8bf*)(&gB[(long)(n0 + r) * p.ldB + k0 + kk]);
      }
    }
    __syncthreads();
    // ---- WMMA compute: 2 k-steps x (2 m-frags x 4 n-frags) ----
#pragma unroll
    for (int ks = 0; ks < KT; ks += 32) {
      FragU a[2], b[4];
#pragma unroll
      for (int mf = 0; mf < 2; ++mf) {
        // A frag: lanes 0-15 K {0..7,16..23}, lanes 16-31 K {8..15,24..31}
        const bf16* ap = &shA[(wm * 32 + mf * 16 + l15) * LDW + ks + hi * 8];
        a[mf].h[0] = *(const v8bf*)(ap);
        a[mf].h[1] = *(const v8bf*)(ap + 16);
      }
#pragma unroll
      for (int nf = 0; nf < 4; ++nf) {
        // B frag: lanes 0-15 K {0..7,16..23}, lanes 16-31 K {8..15,24..31}
        const bf16* bp = &shB[(wn * 64 + nf * 16 + l15) * LDW + ks + hi * 8];
        b[nf].h[0] = *(const v8bf*)(bp);
        b[nf].h[1] = *(const v8bf*)(bp + 16);
      }
#pragma unroll
      for (int mf = 0; mf < 2; ++mf)
#pragma unroll
        for (int nf = 0; nf < 4; ++nf)
          acc[mf][nf] = __builtin_amdgcn_wmma_f32_16x16x32_bf16(
              false, a[mf].v, false, b[nf].v, (short)0, acc[mf][nf],
              false, false);
    }
  }

  // ---- epilogue (C layout: VGPR r -> m = r + 8*hi, lane -> n) ----
#pragma unroll
  for (int mf = 0; mf < 2; ++mf)
#pragma unroll
    for (int nf = 0; nf < 4; ++nf) {
      const int mBase = m0 + wm * 32 + mf * 16 + hi * 8;
      const int n     = n0 + wn * 64 + nf * 16 + l15;
#pragma unroll
      for (int r = 0; r < 8; ++r) {
        const float v = acc[mf][nf][r];
        const int   m = mBase + r;
        if (EP == EP_SCALED_T) {
          p.Of[offO + (long)n * p.ldO + m] = v * p.scale;
        } else if (EP == EP_BF16) {
          p.Ob[offO + (long)m * p.ldO + n] = (bf16)v;
        } else if (EP == EP_RELU_RES) {
          const long idx = (long)m * p.ldO + n;
          p.Of[offO + idx] = fmaxf(v, 0.f) + p.Res[offR + idx];
        } else {  // EP_RES
          const long idx = (long)m * p.ldO + n;
          p.Of[offO + idx] = v + p.Res[offR + idx];
        }
      }
    }
}

// ---------------- elementwise / reduction kernels -------------------------

// out[b,c,l] = x + sin(l*freq(c) + phase(c)); one block per (b,c) row
__global__ __launch_bounds__(256) void peadd_k(const float* __restrict__ x,
                                               float* __restrict__ out) {
  const int bc = blockIdx.x;
  const int c  = bc & 511;
  const int ce = c & ~1;
  float f = __expf(-(float)ce * (9.2103403719761836f / 512.f));  // 10000^(-ce/C)
  float ph = 0.f;
  if (c & 1) { f = -f; ph = 1.5707963267948966f; }
  const long base = (long)bc * 1024;
#pragma unroll
  for (int j = 0; j < 4; ++j) {
    const int l = threadIdx.x + j * 256;
    out[base + l] = x[base + l] + __sinf((float)l * f + ph);
  }
}

// row-wise norm over L=1024: (x-mean)/(sqrt(var_{n-1})+eps); f32 or bf16 out
__global__ __launch_bounds__(256) void norm_k(const float* __restrict__ in,
                                              float* __restrict__ outF,
                                              bf16* __restrict__ outB,
                                              int toBf16) {
  __shared__ float2 red[256];
  const int tid = threadIdx.x;
  const long base = (long)blockIdx.x * 1024;
  const float4 v = *(const float4*)(in + base + tid * 4);
  float s  = v.x + v.y + v.z + v.w;
  float ss = v.x * v.x + v.y * v.y + v.z * v.z + v.w * v.w;
  red[tid] = make_float2(s, ss);
  __syncthreads();
  for (int o = 128; o > 0; o >>= 1) {
    if (tid < o) {
      red[tid].x += red[tid + o].x;
      red[tid].y += red[tid + o].y;
    }
    __syncthreads();
  }
  const float mean = red[0].x * (1.f / 1024.f);
  const float var  = (red[0].y - 1024.f * mean * mean) * (1.f / 1023.f);
  const float inv  = 1.f / (sqrtf(fmaxf(var, 0.f)) + 1e-6f);
  if (toBf16) {
    bf16* op = outB + base + tid * 4;
    op[0] = (bf16)((v.x - mean) * inv);
    op[1] = (bf16)((v.y - mean) * inv);
    op[2] = (bf16)((v.z - mean) * inv);
    op[3] = (bf16)((v.w - mean) * inv);
  } else {
    float4 o;
    o.x = (v.x - mean) * inv; o.y = (v.y - mean) * inv;
    o.z = (v.z - mean) * inv; o.w = (v.w - mean) * inv;
    *(float4*)(outF + base + tid * 4) = o;
  }
}

// depthwise conv1d K=7, pad 3, per-channel weights; bf16 output for GEMM
__global__ __launch_bounds__(256) void dwconv_k(const float* __restrict__ xn,
                                                const float* __restrict__ dw,
                                                bf16* __restrict__ out,
                                                int iconv) {
  const int bc = blockIdx.x;
  const int c  = bc & 511;
  const float* w = dw + ((long)iconv * 512 + c) * 7;
  float wr[7];
#pragma unroll
  for (int t = 0; t < 7; ++t) wr[t] = w[t];
  const long base = (long)bc * 1024;
#pragma unroll
  for (int j = 0; j < 4; ++j) {
    const int l = threadIdx.x + j * 256;
    float a = 0.f;
#pragma unroll
    for (int t = 0; t < 7; ++t) {
      const int pidx = l + t - 3;
      if ((unsigned)pidx < 1024u) a += wr[t] * xn[base + pidx];
    }
    out[base + l] = (bf16)a;
  }
}

// softmax over contiguous row of 1024 (scores stored [l][m]); bf16 out
__global__ __launch_bounds__(256) void softmax_k(const float* __restrict__ S,
                                                 bf16* __restrict__ O) {
  __shared__ float red[256];
  const int tid = threadIdx.x;
  const long base = (long)blockIdx.x * 1024;
  const float4 v = *(const float4*)(S + base + tid * 4);
  red[tid] = fmaxf(fmaxf(v.x, v.y), fmaxf(v.z, v.w));
  __syncthreads();
  for (int o = 128; o > 0; o >>= 1) {
    if (tid < o) red[tid] = fmaxf(red[tid], red[tid + o]);
    __syncthreads();
  }
  const float M = red[0];
  __syncthreads();
  const float e0 = __expf(v.x - M), e1 = __expf(v.y - M);
  const float e2 = __expf(v.z - M), e3 = __expf(v.w - M);
  red[tid] = e0 + e1 + e2 + e3;
  __syncthreads();
  for (int o = 128; o > 0; o >>= 1) {
    if (tid < o) red[tid] += red[tid + o];
    __syncthreads();
  }
  const float inv = 1.f / red[0];
  bf16* op = O + base + tid * 4;
  op[0] = (bf16)(e0 * inv); op[1] = (bf16)(e1 * inv);
  op[2] = (bf16)(e2 * inv); op[3] = (bf16)(e3 * inv);
}

// elementwise f32 -> bf16 (weight conversion)
__global__ __launch_bounds__(256) void f2b_k(const float* __restrict__ in,
                                             bf16* __restrict__ out, long n) {
  for (long i = (long)blockIdx.x * 256 + threadIdx.x; i < n;
       i += (long)gridDim.x * 256)
    out[i] = (bf16)in[i];
}

// ---------------------------- host side -----------------------------------

static inline GemmP basep() {
  GemmP p{};
  p.mA = p.mB = p.mO = p.mR = (1 << 30);
  p.scale = 1.f;
  return p;
}

extern "C" void kernel_launch(void* const* d_in, const int* in_sizes, int n_in,
                              void* d_out, int out_size, void* d_ws,
                              size_t ws_size, hipStream_t stream) {
  (void)in_sizes; (void)n_in; (void)out_size;
  const float* x  = (const float*)d_in[0];
  const float* dw = (const float*)d_in[1];
  const float* pw = (const float*)d_in[2];
  const float* Wq = (const float*)d_in[3];
  const float* Wk = (const float*)d_in[4];
  const float* Wv = (const float*)d_in[5];
  const float* Wo = (const float*)d_in[6];
  const float* W  = (const float*)d_in[7];
  float* out = (float*)d_out;

  const long QKVW = 8L * 16 * 64 * 512;   // 4,194,304
  const long SQW  = 16L * 512 * 512;      // 4,194,304 (Wo, W)
  const long PWW  = 4L * 512 * 512;       // 1,048,576
  const long BCL  = 16L * 512 * 1024;     // 8,388,608
  const long HBDL = 128L * 64 * 1024;     // 8,388,608

  char* ws = (char*)d_ws;
  size_t off = 0;
  auto alloc = [&](size_t bytes) -> char* {
    char* p = ws + off;
    off += (bytes + 255) & ~(size_t)255;
    return p;
  };

  bf16* wqB = (bf16*)alloc(QKVW * 2);
  bf16* wkB = (bf16*)alloc(QKVW * 2);
  bf16* wvB = (bf16*)alloc(QKVW * 2);
  bf16* woB = (bf16*)alloc(SQW * 2);
  bf16* wB  = (bf16*)alloc(SQW * 2);
  bf16* pwB = (bf16*)alloc(PWW * 2);
  float* cur   = (float*)alloc(BCL * 4);   // running activation / residual
  float* normf = (float*)alloc(BCL * 4);   // f32 norm (feeds depthwise conv)
  bf16* dwb    = (bf16*)alloc(BCL * 2);    // depthwise out, bf16
  bf16* xnb    = (bf16*)alloc(BCL * 2);    // bf16 norm (attention / FF input)
  bf16* Qb     = (bf16*)alloc(HBDL * 2);
  bf16* Kb     = (bf16*)alloc(HBDL * 2);
  bf16* Vb     = (bf16*)alloc(HBDL * 2);
  bf16* headb  = (bf16*)alloc(BCL * 2);

  // size the attention scores chunk from remaining workspace
  const size_t perB = 1024ull * 1024 * 4 + 1024ull * 1024 * 2 + 1024;
  long avail = (long)ws_size - (long)off;
  int nb = 1;
  if (avail > (long)perB) {
    long t = avail / (long)perB;
    nb = t > 128 ? 128 : (int)t;
  }
  float* Sbuf  = (float*)alloc((size_t)nb * 1024 * 1024 * 4);  // [nb][l][m]
  bf16*  attnT = (bf16*)alloc((size_t)nb * 1024 * 1024 * 2);   // [nb][l][m]

  // ---- convert weights to bf16 ----
  f2b_k<<<2048, 256, 0, stream>>>(Wq, wqB, QKVW);
  f2b_k<<<2048, 256, 0, stream>>>(Wk, wkB, QKVW);
  f2b_k<<<2048, 256, 0, stream>>>(Wv, wvB, QKVW);
  f2b_k<<<2048, 256, 0, stream>>>(Wo, woB, SQW);
  f2b_k<<<2048, 256, 0, stream>>>(W,  wB,  SQW);
  f2b_k<<<2048, 256, 0, stream>>>(pw, pwB, PWW);

  // ---- positional encoding ----
  peadd_k<<<8192, 256, 0, stream>>>(x, cur);

  // ---- 4x [norm -> depthwise -> pointwise GEMM(+relu+res)] ----
  for (int i = 0; i < 4; ++i) {
    norm_k<<<8192, 256, 0, stream>>>(cur, normf, nullptr, 0);
    dwconv_k<<<8192, 256, 0, stream>>>(normf, dw, dwb, i);
    GemmP p = basep();
    p.A = pwB + (long)i * 512 * 512; p.sA = 0;
    p.B = dwb;  p.sB = 512L * 1024;
    p.Res = cur; p.sR = 512L * 1024;
    p.Of = cur;  p.sO = 512L * 1024;
    p.ldA = 512; p.ldB = 1024; p.ldO = 1024; p.Kdim = 512;
    gemm_k<128, 128, 4, 2, 0, 0, EP_RELU_RES>
        <<<dim3(8, 4, 16), 256, 0, stream>>>(p);
  }

  // ---- attention: norm -> QKV -> scores -> softmax -> V*A -> Wo(+res) ----
  norm_k<<<8192, 256, 0, stream>>>(cur, nullptr, xnb, 1);

  const bf16* wqkv[3] = {wqB, wkB, wvB};
  bf16* qkv[3] = {Qb, Kb, Vb};
  for (int t = 0; t < 3; ++t) {
    GemmP p = basep();
    p.A = wqkv[t]; p.sA = 64L * 512;
    p.B = xnb; p.sB = 512L * 1024; p.mB = 16;   // b = z % 16 (z = h*16+b)
    p.Ob = qkv[t]; p.sO = 64L * 1024;
    p.ldA = 512; p.ldB = 1024; p.ldO = 1024; p.Kdim = 512;
    gemm_k<64, 256, 2, 4, 0, 0, EP_BF16>
        <<<dim3(4, 1, 128), 256, 0, stream>>>(p);
  }

  for (int z0 = 0; z0 < 128; z0 += nb) {
    const int cnt = (128 - z0) < nb ? (128 - z0) : nb;
    {  // scores: S^T[l][m] = (Q^T K)[m,l] / 8, per (h,b)
      GemmP p = basep();
      p.A = Qb; p.sA = 64L * 1024; p.zA = z0; p.ldA = 1024;  // [K=64][M]
      p.B = Kb; p.sB = 64L * 1024; p.zB = z0; p.ldB = 1024;  // [K=64][N]
      p.Of = Sbuf; p.sO = 1024L * 1024; p.zO = 0; p.ldO = 1024;
      p.Kdim = 64; p.scale = 0.125f;
      gemm_k<128, 128, 4, 2, 1, 0, EP_SCALED_T>
          <<<dim3(8, 8, cnt), 256, 0, stream>>>(p);
    }
    softmax_k<<<cnt * 1024, 256, 0, stream>>>(Sbuf, attnT);
    {  // head[d,l] = V[d,m] * A[m,l]; attnT is [l][m] => direct [N][K] tile
      GemmP p = basep();
      p.A = Vb; p.sA = 64L * 1024; p.zA = z0; p.ldA = 1024;
      p.B = attnT; p.sB = 1024L * 1024; p.zB = 0; p.ldB = 1024;
      p.Ob = headb; p.headOut = 1; p.zO = z0; p.ldO = 1024;
      p.Kdim = 1024;
      gemm_k<64, 256, 2, 4, 0, 1, EP_BF16>
          <<<dim3(4, 1, cnt), 256, 0, stream>>>(p);
    }
  }

  {  // Wo projection + residual (no relu)
    GemmP p = basep();
    p.A = woB; p.sA = 512L * 512;
    p.B = headb; p.sB = 512L * 1024;
    p.Res = cur; p.sR = 512L * 1024;
    p.Of = cur;  p.sO = 512L * 1024;
    p.ldA = 512; p.ldB = 1024; p.ldO = 1024; p.Kdim = 512;
    gemm_k<128, 128, 4, 2, 0, 0, EP_RES>
        <<<dim3(8, 4, 16), 256, 0, stream>>>(p);
  }

  // ---- feed-forward: relu(W @ norm(cur)) + cur -> d_out ----
  norm_k<<<8192, 256, 0, stream>>>(cur, nullptr, xnb, 1);
  {
    GemmP p = basep();
    p.A = wB; p.sA = 512L * 512;
    p.B = xnb; p.sB = 512L * 1024;
    p.Res = cur; p.sR = 512L * 1024;
    p.Of = out;  p.sO = 512L * 1024;
    p.ldA = 512; p.ldB = 1024; p.ldO = 1024; p.Kdim = 512;
    gemm_k<128, 128, 4, 2, 0, 0, EP_RELU_RES>
        <<<dim3(8, 4, 16), 256, 0, stream>>>(p);
  }
}
